// MoELayer_70720931496632
// MI455X (gfx1250) — matrix operand order repaired and checked by reading, not compiled, not run on hardware
//
#include <hip/hip_runtime.h>

// ---------------------------------------------------------------------------
// MoE (BASE layer) forward for MI455X / gfx1250.
// gate -> bucket per expert -> LN(bf16, bucket order) -> grouped WMMA GEMMs.
// GEMM: 64x128 block tile, 8 waves, wave = 16x64 (4x v_wmma_f32_16x16x32_bf16
// per K-step, shared A frag, back-to-back issue). Weights pre-transposed to
// [E][N][K] bf16 so LDS staging is contiguous; staging uses
// global_load_async_to_lds_b128 with double-buffered LDS (XOR-toggled
// offsets, unroll-disabled loop to keep accumulators register-stable).
// ---------------------------------------------------------------------------

#define SDIM 2048
#define BDIM 2
#define DDIM 1024
#define FDIM 4096
#define EDIM 8
#define TTOK 4096            // SDIM * BDIM
#define LN_EPS 1e-5f

typedef __attribute__((ext_vector_type(16))) __bf16 bf16x16;
typedef __attribute__((ext_vector_type(8)))  float  f32x8;

union FragU { uint4 q[2]; bf16x16 v; };

// token t (= b*S + s) -> flat offset into [S,B,D] tensor
__device__ __forceinline__ size_t xoff(int t) {
  return (size_t)(t & (SDIM - 1)) * (BDIM * DDIM) + (size_t)(t >> 11) * DDIM;
}

// fp32 -> bf16 (round to nearest even)
__device__ __forceinline__ unsigned short f2bf(float f) {
  unsigned u = __float_as_uint(f);
  unsigned r = u + 0x7FFFu + ((u >> 16) & 1u);
  return (unsigned short)(r >> 16);
}

// async DMA: 16 bytes global -> LDS (per lane), tracked by ASYNCcnt
__device__ __forceinline__ void async_b128(unsigned ldsOff, const void* gaddr) {
  asm volatile("global_load_async_to_lds_b128 %0, %1, off"
               :: "v"(ldsOff), "v"(gaddr) : "memory");
}
__device__ __forceinline__ void wait_async_le3() {
  asm volatile("s_wait_asynccnt 0x3" ::: "memory");
}
__device__ __forceinline__ void wait_async_0() {
  asm volatile("s_wait_asynccnt 0x0" ::: "memory");
}

// ---------------------------------------------------------------------------
__global__ void init_kernel(int* counts) {
  if (threadIdx.x < EDIM) counts[threadIdx.x] = 0;
}

// W [E][KD][ND] fp32 -> WT [E][ND][KD] bf16, 64x64 tiles via LDS transpose
template <int KD, int ND>
__global__ __launch_bounds__(256)
void cvtT_kernel(const float* __restrict__ W, unsigned short* __restrict__ WT) {
  __shared__ __align__(16) unsigned short tile[64][72];   // 144B row stride
  const int e  = blockIdx.z;
  const int n0 = blockIdx.x * 64;
  const int k0 = blockIdx.y * 64;
  const float* We = W + (size_t)e * KD * ND;
  unsigned short* WTe = WT + (size_t)e * ND * KD;
  const int tid = threadIdx.x;
  const int kr = tid >> 2, seg = tid & 3;                 // 64 k-rows, 4 thr/row
  const float* src = We + (size_t)(k0 + kr) * ND + n0 + seg * 16;
#pragma unroll
  for (int u = 0; u < 4; u++) {
    float4 f = *(const float4*)(src + u * 4);
    int n = seg * 16 + u * 4;
    tile[n + 0][kr] = f2bf(f.x);
    tile[n + 1][kr] = f2bf(f.y);
    tile[n + 2][kr] = f2bf(f.z);
    tile[n + 3][kr] = f2bf(f.w);
  }
  __syncthreads();
  const int nr = tid >> 2;                                // 64 n-rows, 16 k/thr
  unsigned short* dst = WTe + (size_t)(n0 + nr) * KD + k0 + seg * 16;
  uint4 a = *(const uint4*)&tile[nr][seg * 16];
  uint4 b = *(const uint4*)&tile[nr][seg * 16 + 8];
  *(uint4*)(dst) = a;
  *(uint4*)(dst + 8) = b;
}

// one wave per token: 8 dot products of length D, shuffle-reduce, argmax
__global__ void gate_kernel(const float* __restrict__ x,
                            const float* __restrict__ cent,
                            int* __restrict__ assign, int* __restrict__ counts) {
  int wid = (blockIdx.x * blockDim.x + threadIdx.x) >> 5;
  int lane = threadIdx.x & 31;
  if (wid >= TTOK) return;
  const float* xr = x + xoff(wid);
  float p[EDIM];
#pragma unroll
  for (int e = 0; e < EDIM; e++) p[e] = 0.f;
#pragma unroll
  for (int j = 0; j < 8; j++) {
    float4 xv = *(const float4*)(xr + j * 128 + lane * 4);
#pragma unroll
    for (int e = 0; e < EDIM; e++) {
      float4 cv = *(const float4*)(cent + e * DDIM + j * 128 + lane * 4);
      p[e] += xv.x * cv.x + xv.y * cv.y + xv.z * cv.z + xv.w * cv.w;
    }
  }
#pragma unroll
  for (int e = 0; e < EDIM; e++) {
    float v = p[e];
#pragma unroll
    for (int off = 16; off > 0; off >>= 1) v += __shfl_xor(v, off);
    p[e] = v;
  }
  if (lane == 0) {
    int bi = 0; float bv = p[0];
#pragma unroll
    for (int e = 1; e < EDIM; e++) if (p[e] > bv) { bv = p[e]; bi = e; }
    assign[wid] = bi;
    atomicAdd(&counts[bi], 1);
  }
}

__global__ void scan_kernel(const int* counts, int* offs, int* cursor) {
  if (threadIdx.x == 0) {
    int o = 0;
    for (int e = 0; e < EDIM; e++) { offs[e] = o; cursor[e] = o; o += counts[e]; }
  }
}

__global__ void scatter_kernel(const int* __restrict__ assign,
                               int* __restrict__ cursor, int* __restrict__ bucket) {
  int t = blockIdx.x * blockDim.x + threadIdx.x;
  if (t >= TTOK) return;
  int e = assign[t];
  int p = atomicAdd(&cursor[e], 1);
  bucket[p] = t;
}

// one wave per bucket row: LN with expert params, write bf16 in bucket order
__global__ void ln_kernel(const float* __restrict__ x,
                          const float* __restrict__ g, const float* __restrict__ b,
                          const int* __restrict__ bucket, const int* __restrict__ assign,
                          unsigned short* __restrict__ xnb) {
  int wid = (blockIdx.x * blockDim.x + threadIdx.x) >> 5;
  int lane = threadIdx.x & 31;
  if (wid >= TTOK) return;
  int t = bucket[wid];
  int e = assign[t];
  const float* xr = x + xoff(t);
  float4 v[8];
  float s = 0.f, s2 = 0.f;
#pragma unroll
  for (int j = 0; j < 8; j++) {
    v[j] = *(const float4*)(xr + j * 128 + lane * 4);
    s  += v[j].x + v[j].y + v[j].z + v[j].w;
    s2 += v[j].x * v[j].x + v[j].y * v[j].y + v[j].z * v[j].z + v[j].w * v[j].w;
  }
#pragma unroll
  for (int off = 16; off > 0; off >>= 1) { s += __shfl_xor(s, off); s2 += __shfl_xor(s2, off); }
  float mu  = s * (1.0f / DDIM);
  float var = s2 * (1.0f / DDIM) - mu * mu;
  float rs  = rsqrtf(var + LN_EPS);
  const float* gr = g + (size_t)e * DDIM;
  const float* br = b + (size_t)e * DDIM;
  unsigned short* orow = xnb + (size_t)wid * DDIM;
#pragma unroll
  for (int j = 0; j < 8; j++) {
    int d = j * 128 + lane * 4;
    float4 gv = *(const float4*)(gr + d);
    float4 bv = *(const float4*)(br + d);
    float r0 = (v[j].x - mu) * rs * gv.x + bv.x;
    float r1 = (v[j].y - mu) * rs * gv.y + bv.y;
    float r2 = (v[j].z - mu) * rs * gv.z + bv.z;
    float r3 = (v[j].w - mu) * rs * gv.w + bv.w;
    uint2 pk;
    pk.x = (unsigned)f2bf(r0) | ((unsigned)f2bf(r1) << 16);
    pk.y = (unsigned)f2bf(r2) | ((unsigned)f2bf(r3) << 16);
    *(uint2*)(orow + d) = pk;
  }
}

// ---------------------------------------------------------------------------
// Grouped GEMM: block tile 64(rows) x 128(cols), K-step 32, 8 waves.
// Waves: 4 row-groups x 2 col-groups; each wave: 16 rows x 64 cols = 4 C tiles.
// Double-buffered LDS via XOR-toggled byte offsets; K loop unroll-disabled and
// last step peeled so the body is branchless and acc registers stay fixed.
// All 4 B fragments loaded before the 4 back-to-back WMMAs.
// ---------------------------------------------------------------------------
template <bool FF1, int KD, int ND>
__global__ __launch_bounds__(256)
void gemm_kernel(const unsigned short* __restrict__ A,    // [rows_padded, KD]
                 const unsigned short* __restrict__ BT,   // [E, ND, KD]
                 const float* __restrict__ bias,          // [E, ND]
                 const int* __restrict__ counts, const int* __restrict__ offs,
                 const int* __restrict__ bucket,
                 const float* __restrict__ xin,           // residual (FF2)
                 unsigned short* __restrict__ Hout,       // FF1 output
                 float* __restrict__ Yout)                // FF2 output (d_out)
{
  const int e = blockIdx.z;
  const int cnt = counts[e];
  const int row0 = blockIdx.y * 64;
  if (row0 >= cnt) return;                   // ragged per-expert bound
  const int grow0 = offs[e] + row0;
  const int n0 = blockIdx.x * 128;

  __shared__ __align__(16) unsigned short lA[2][64][40];   // 80B row stride
  __shared__ __align__(16) unsigned short lB[2][128][40];
  const unsigned ASTRIDE = 64 * 40 * 2;      // 5120 B per A buffer
  const unsigned BSTRIDE = 128 * 40 * 2;     // 10240 B per B buffer

  const int tid = threadIdx.x;
  const int lane = tid & 31;
  const int w = tid >> 5;
  const int wr = w & 3;                      // 4 row groups of 16
  const int wc = w >> 2;                     // 2 col groups of 64
  const int half = lane >> 4, l15 = lane & 15;

  const unsigned short* BTe = BT + (size_t)e * ND * KD;

  // cooperative stage: A: 64 rows x 32k (16B/thr), B: 128 rows x 32k (32B/thr)
  const unsigned short* Ap = A + (size_t)(grow0 + (tid >> 2)) * KD + (tid & 3) * 8;
  const unsigned short* Bp = BTe + (size_t)(n0 + (tid >> 1)) * KD + (tid & 1) * 16;
  const unsigned stA0 = (unsigned)(size_t)&lA[0][tid >> 2][(tid & 3) * 8];
  const unsigned stB0 = (unsigned)(size_t)&lB[0][tid >> 1][(tid & 1) * 16];

  // per-wave fragment read offsets (within one buffer)
  const char* lAb = (const char*)&lA[0][0][0] + (unsigned)((wr * 16 + l15) * 80 + half * 32);
  const char* lBb = (const char*)&lB[0][0][0] + (unsigned)((wc * 64 + l15) * 80 + half * 32);

  f32x8 acc[4] = {};

  // prologue: stage K-step 0 into buffer 0
  async_b128(stA0, Ap);
  async_b128(stB0, Bp);
  async_b128(stB0 + 16, Bp + 8);

  unsigned rdA = 0, rdB = 0;                 // read-buffer byte offsets
  unsigned wrA = stA0 + ASTRIDE, wrB = stB0 + BSTRIDE;  // write side (buf 1)
  const unsigned short* apG = Ap + 32;
  const unsigned short* bpG = Bp + 32;

  const int NK = KD / 32;
#pragma clang loop unroll(disable)
  for (int i = 0; i < NK - 1; i++) {
    // issue next K-step into the other buffer (overlaps with compute below)
    async_b128(wrA, apG);
    async_b128(wrB, bpG);
    async_b128(wrB + 16, bpG + 8);
    apG += 32; bpG += 32;
    wait_async_le3();                        // oldest 3 (current step) landed
    __syncthreads();                         // all waves' tiles visible

    FragU fa, fb0, fb1, fb2, fb3;
    fa.q[0]  = *(const uint4*)(lAb + rdA);
    fa.q[1]  = *(const uint4*)(lAb + rdA + 16);
    fb0.q[0] = *(const uint4*)(lBb + rdB);
    fb0.q[1] = *(const uint4*)(lBb + rdB + 16);
    fb1.q[0] = *(const uint4*)(lBb + rdB + 1280);
    fb1.q[1] = *(const uint4*)(lBb + rdB + 1296);
    fb2.q[0] = *(const uint4*)(lBb + rdB + 2560);
    fb2.q[1] = *(const uint4*)(lBb + rdB + 2576);
    fb3.q[0] = *(const uint4*)(lBb + rdB + 3840);
    fb3.q[1] = *(const uint4*)(lBb + rdB + 3856);
    acc[0] = __builtin_amdgcn_wmma_f32_16x16x32_bf16(false, fa.v, false, fb0.v, (short)0, acc[0], false, false);
    acc[1] = __builtin_amdgcn_wmma_f32_16x16x32_bf16(false, fa.v, false, fb1.v, (short)0, acc[1], false, false);
    acc[2] = __builtin_amdgcn_wmma_f32_16x16x32_bf16(false, fa.v, false, fb2.v, (short)0, acc[2], false, false);
    acc[3] = __builtin_amdgcn_wmma_f32_16x16x32_bf16(false, fa.v, false, fb3.v, (short)0, acc[3], false, false);

    __syncthreads();                         // tiles consumed -> buffer reusable
    rdA ^= ASTRIDE; rdB ^= BSTRIDE;
    wrA ^= ASTRIDE; wrB ^= BSTRIDE;
  }

  // peeled last K-step (no prefetch)
  wait_async_0();
  __syncthreads();
  {
    FragU fa, fb0, fb1, fb2, fb3;
    fa.q[0]  = *(const uint4*)(lAb + rdA);
    fa.q[1]  = *(const uint4*)(lAb + rdA + 16);
    fb0.q[0] = *(const uint4*)(lBb + rdB);
    fb0.q[1] = *(const uint4*)(lBb + rdB + 16);
    fb1.q[0] = *(const uint4*)(lBb + rdB + 1280);
    fb1.q[1] = *(const uint4*)(lBb + rdB + 1296);
    fb2.q[0] = *(const uint4*)(lBb + rdB + 2560);
    fb2.q[1] = *(const uint4*)(lBb + rdB + 2576);
    fb3.q[0] = *(const uint4*)(lBb + rdB + 3840);
    fb3.q[1] = *(const uint4*)(lBb + rdB + 3856);
    acc[0] = __builtin_amdgcn_wmma_f32_16x16x32_bf16(false, fa.v, false, fb0.v, (short)0, acc[0], false, false);
    acc[1] = __builtin_amdgcn_wmma_f32_16x16x32_bf16(false, fa.v, false, fb1.v, (short)0, acc[1], false, false);
    acc[2] = __builtin_amdgcn_wmma_f32_16x16x32_bf16(false, fa.v, false, fb2.v, (short)0, acc[2], false, false);
    acc[3] = __builtin_amdgcn_wmma_f32_16x16x32_bf16(false, fa.v, false, fb3.v, (short)0, acc[3], false, false);
  }

#pragma unroll
  for (int j = 0; j < 4; j++) {
    const int ncol = n0 + wc * 64 + j * 16 + l15;
    const float bv = bias[(size_t)e * ND + ncol];
#pragma unroll
    for (int r = 0; r < 8; r++) {
      int rloc = wr * 16 + half * 8 + r;     // C layout: VGPR r -> M=r(+8), N=lane
      if (row0 + rloc < cnt) {
        float v = acc[j][r] + bv;
        if (FF1) {
          v = v > 0.f ? v : 0.f;
          Hout[(size_t)(grow0 + rloc) * ND + ncol] = f2bf(v);
        } else {
          int t = bucket[grow0 + rloc];
          size_t xo = xoff(t) + ncol;
          Yout[xo] = v + xin[xo];
        }
      }
    }
  }
}

// ---------------------------------------------------------------------------
extern "C" void kernel_launch(void* const* d_in, const int* in_sizes, int n_in,
                              void* d_out, int out_size, void* d_ws, size_t ws_size,
                              hipStream_t stream) {
  const float* x    = (const float*)d_in[0];  // [S,B,D]
  const float* cent = (const float*)d_in[1];  // [E,D]
  const float* ln_g = (const float*)d_in[2];  // [E,D]
  const float* ln_b = (const float*)d_in[3];  // [E,D]
  const float* W1   = (const float*)d_in[4];  // [E,D,F]
  const float* b1   = (const float*)d_in[5];  // [E,F]
  const float* W2   = (const float*)d_in[6];  // [E,F,D]
  const float* b2   = (const float*)d_in[7];  // [E,D]
  float* out = (float*)d_out;

  char* ws = (char*)d_ws;
  size_t off = 0;
  auto take = [&](size_t bytes) -> size_t {
    size_t o = off;
    off += (bytes + 255) & ~(size_t)255;
    return o;
  };
  int* counts = (int*)(ws + take(EDIM * 4));
  int* offs   = (int*)(ws + take(EDIM * 4));
  int* cursor = (int*)(ws + take(EDIM * 4));
  int* assign = (int*)(ws + take((size_t)TTOK * 4));
  int* bucket = (int*)(ws + take((size_t)TTOK * 4));
  unsigned short* xnb  = (unsigned short*)(ws + take((size_t)(TTOK + 64) * DDIM * 2));
  unsigned short* h    = (unsigned short*)(ws + take((size_t)(TTOK + 64) * FDIM * 2));
  unsigned short* W1bt = (unsigned short*)(ws + take((size_t)EDIM * DDIM * FDIM * 2));
  unsigned short* W2bt = (unsigned short*)(ws + take((size_t)EDIM * FDIM * DDIM * 2));
  (void)ws_size; (void)in_sizes; (void)n_in; (void)out_size;

  init_kernel<<<1, 32, 0, stream>>>(counts);

  // W1 [E,D,F] -> W1bt [E,F,D] bf16 ; W2 [E,F,D] -> W2bt [E,D,F] bf16
  cvtT_kernel<DDIM, FDIM><<<dim3(FDIM / 64, DDIM / 64, EDIM), 256, 0, stream>>>(W1, W1bt);
  cvtT_kernel<FDIM, DDIM><<<dim3(DDIM / 64, FDIM / 64, EDIM), 256, 0, stream>>>(W2, W2bt);

  gate_kernel<<<(TTOK * 32) / 256, 256, 0, stream>>>(x, cent, assign, counts);
  scan_kernel<<<1, 32, 0, stream>>>(counts, offs, cursor);
  scatter_kernel<<<TTOK / 256, 256, 0, stream>>>(assign, cursor, bucket);
  ln_kernel<<<(TTOK * 32) / 256, 256, 0, stream>>>(x, ln_g, ln_b, bucket, assign, xnb);

  // ff1: [cnt_e, D] x [D, F] -> relu -> h (bf16)
  gemm_kernel<true, DDIM, FDIM><<<dim3(FDIM / 128, TTOK / 64, EDIM), 256, 0, stream>>>(
      xnb, W1bt, b1, counts, offs, bucket, nullptr, h, nullptr);
  // ff2: [cnt_e, F] x [F, D] + b2 + residual -> d_out (fp32)
  gemm_kernel<false, FDIM, DDIM><<<dim3(DDIM / 128, TTOK / 64, EDIM), 256, 0, stream>>>(
      h, W2bt, b2, counts, offs, bucket, x, nullptr, out);
}
